// Lstm_69630009803325
// MI455X (gfx1250) — matrix operand order repaired and checked by reading, not compiled, not run on hardware
//
#include <hip/hip_runtime.h>
#include <hip/hip_bf16.h>

// ---- shapes ----
#define B_  64
#define T_  256
#define I_  512
#define H_  1024
#define G4H 4096   // 4*H

typedef __attribute__((ext_vector_type(16))) __bf16 v16bf;
typedef __attribute__((ext_vector_type(8)))  __bf16 v8bf;
typedef __attribute__((ext_vector_type(8)))  float  v8f;
typedef __attribute__((ext_vector_type(4)))  unsigned int v4u;
typedef __attribute__((ext_vector_type(8)))  int v8i;
typedef __attribute__((ext_vector_type(4)))  int v4i;

#if defined(__has_builtin)
#  if __has_builtin(__builtin_amdgcn_tensor_load_to_lds) && \
      __has_builtin(__builtin_amdgcn_s_wait_tensorcnt)
#    define USE_TDM 1
#  else
#    define USE_TDM 0
#  endif
#else
#  define USE_TDM 0
#endif

static __device__ __forceinline__ v16bf cat8(v8bf lo, v8bf hi) {
  return __builtin_shufflevector(lo, hi, 0,1,2,3,4,5,6,7,8,9,10,11,12,13,14,15);
}
static __device__ __forceinline__ v8bf ldv8(const __bf16* p) {
  return *reinterpret_cast<const v8bf*>(p);
}
static __device__ __forceinline__ v8f zero8() {
  v8f z = {0.f,0.f,0.f,0.f,0.f,0.f,0.f,0.f};
  return z;
}
static __device__ __forceinline__ float sigmoidf_(float x) {
  return 1.0f / (1.0f + __expf(-x));
}

// x: [B,T,I] fp32  ->  xbf: [T*B, I] bf16  (row r = t*B + b)
__global__ void k_convert_x(const float* __restrict__ x, __bf16* __restrict__ xbf) {
  long long total = (long long)B_ * T_ * I_;
  long long stride = (long long)gridDim.x * blockDim.x;
  for (long long idx = (long long)blockIdx.x * blockDim.x + threadIdx.x;
       idx < total; idx += stride) {
    int k  = (int)(idx % I_);
    int bt = (int)(idx / I_);
    int b  = bt / T_;
    int t  = bt % T_;
    xbf[((long long)(t * B_ + b)) * I_ + k] = (__bf16)x[idx];
  }
}

// W: [Kin, 4H] fp32 -> WT: [4H, Kin] bf16
__global__ void k_transposeW(const float* __restrict__ W, __bf16* __restrict__ WT, int Kin) {
  long long total = (long long)Kin * G4H;
  long long stride = (long long)gridDim.x * blockDim.x;
  for (long long idx = (long long)blockIdx.x * blockDim.x + threadIdx.x;
       idx < total; idx += stride) {
    int n = (int)(idx % G4H);
    int k = (int)(idx / G4H);
    WT[(long long)n * Kin + k] = (__bf16)W[idx];
  }
}

__global__ void k_init_state(const float* __restrict__ h0l, const float* __restrict__ c0l,
                             __bf16* __restrict__ hbf0, float* __restrict__ cbuf) {
  int idx = blockIdx.x * blockDim.x + threadIdx.x;
  if (idx < B_ * H_) {
    hbf0[idx] = (__bf16)h0l[idx];
    cbuf[idx] = c0l[idx];
  }
}

// C[M,4H] = A[M,K] * BT[4H,K]^T + bias   (bf16 in, fp32 out, WMMA 16x16x32)
// Block: 256 threads (8 waves), WG tile 128(M) x 64(N); wave tile 32x32 = 2x2 WMMA tiles.
__global__ __launch_bounds__(256) void k_gemm_pre(
    const __bf16* __restrict__ A, const __bf16* __restrict__ BT,
    const float* __restrict__ bias, float* __restrict__ C, int M, int K) {
  const int lane = threadIdx.x & 31;
  const int w    = threadIdx.x >> 5;     // 0..7
  const int half = (lane >> 4) & 1;
  const int l16  = lane & 15;
  const int wm = w & 3, wn = w >> 2;     // 4 x 2 wave grid
  const int mbase = blockIdx.y * 128 + wm * 32;
  const int nbase = blockIdx.x * 64  + wn * 32;

  v8f acc[2][2];
  acc[0][0] = zero8(); acc[0][1] = zero8(); acc[1][0] = zero8(); acc[1][1] = zero8();

  for (int kk = 0; kk < K; kk += 32) {
    v16bf a[2], b[2];
    #pragma unroll
    for (int i = 0; i < 2; ++i) {
      // A fragment: row M = l16, lane half selects K runs [8h..8h+7] and [16+8h..16+8h+7]
      const __bf16* ap = A + (long long)(mbase + i * 16 + l16) * K + kk;
      a[i] = cat8(ldv8(ap + 8 * half), ldv8(ap + 16 + 8 * half));
    }
    #pragma unroll
    for (int j = 0; j < 2; ++j) {
      // B fragment: column N = l16, lane half selects K runs [8h..8h+7] and [16+8h..16+8h+7]
      const __bf16* bp = BT + (long long)(nbase + j * 16 + l16) * K + kk;
      b[j] = cat8(ldv8(bp + 8 * half), ldv8(bp + 16 + 8 * half));
    }
    #pragma unroll
    for (int i = 0; i < 2; ++i)
      #pragma unroll
      for (int j = 0; j < 2; ++j)
        acc[i][j] = __builtin_amdgcn_wmma_f32_16x16x32_bf16(
            false, a[i], false, b[j], (short)0, acc[i][j], false, false);
  }

  #pragma unroll
  for (int i = 0; i < 2; ++i)
    #pragma unroll
    for (int j = 0; j < 2; ++j) {
      int col = nbase + j * 16 + l16;
      float bv = bias[col];
      #pragma unroll
      for (int v = 0; v < 8; ++v) {
        int row = mbase + i * 16 + v + 8 * half;  // C/D layout: VGPR v -> M = v + 8*half
        C[(long long)row * G4H + col] = acc[i][j][v] + bv;
      }
    }
}

// One LSTM timestep: gates = Xpre_t + h_in @ W_hh ; pointwise ; write h/c.
// Grid: H/16 = 64 WGs (16 output columns each). Block: 512 threads = 16 waves.
// Wave w -> gate = w&3, m-tile = w>>2 : one 16x16 WMMA tile, K = 1024 (32 WMMAs).
// h_in (64x1024 bf16 = 128KB) is staged into LDS via the Tensor Data Mover
// (one DMA per WG, TENSORcnt) so the A-fragments come from low-latency DS loads
// and the sequential critical path only touches L2 for the W_hh columns.
__global__ __launch_bounds__(512) void k_rec_step(
    const __bf16* __restrict__ hin,    // [B,H] bf16
    __bf16* __restrict__ hout,         // [B,H] bf16 (other ping-pong buffer)
    const __bf16* __restrict__ WhhT,   // [4H,H] bf16
    const float* __restrict__ Xpre_t,  // [B,4H] fp32 (bias already folded in)
    float* __restrict__ cbuf,          // [B,H] fp32
    __bf16* __restrict__ out_bf,       // layer0: [T*B,H] bf16 seq out (else null)
    float* __restrict__ out_f32,       // layer1: d_out outputs [B,T,H] (else null)
    float* __restrict__ hT, float* __restrict__ cT,
    int t, int is_last)
{
  __shared__ __align__(16) __bf16 hlds[B_][H_];   // 128 KB staged h_in
  __shared__ float ldsG[4][B_][17];               // gate exchange, padded
  const int lane = threadIdx.x & 31;
  const int wv   = threadIdx.x >> 5;   // 0..15
  const int gate = wv & 3;
  const int m    = wv >> 2;            // 0..3 (16 batch rows each)
  const int half = (lane >> 4) & 1;
  const int l16  = lane & 15;
  const int n0   = blockIdx.x * 16;

#if USE_TDM
  if (wv == 0) {
    // Tensor DMA descriptor (D#): 2-D tensor, data_size=8B units.
    // Tile = full h_in: 256 (8B units per row) x 64 rows, stride 256.
    unsigned long long ga = (unsigned long long)(uintptr_t)hin;
    unsigned ldsa = (unsigned)(uintptr_t)(&hlds[0][0]);
    v4u g0;
    g0[0] = 1u;                                        // count=1 (valid user D#)
    g0[1] = ldsa;                                      // lds_addr
    g0[2] = (unsigned)(ga & 0xFFFFFFFFu);              // global_addr[31:0]
    g0[3] = (unsigned)((ga >> 32) & 0x01FFFFFFu)       // global_addr[56:32]
          | (2u << 30);                                // type=2 ("image")
    v8i g1;
    g1[0] = (int)(3u << 16);          // workgroup_mask=0, data_size=3 (8B)
    g1[1] = (int)(256u << 16);        // tensor_dim0[15:0]=256 (bits 63:48)
    g1[2] = (int)(64u  << 16);        // tensor_dim0 hi=0, tensor_dim1[15:0]=64
    g1[3] = (int)(256u << 16);        // tensor_dim1 hi=0, tile_dim0=256
    g1[4] = (int)(64u);               // tile_dim1=64, tile_dim2=0
    g1[5] = (int)(256u);              // tensor_dim0_stride lo = 256
    g1[6] = 0;                        // stride hi, tensor_dim1_stride lo
    g1[7] = 0;
    v4i g2 = {0,0,0,0};
    v4i g3 = {0,0,0,0};
    v8i g4 = {0,0,0,0,0,0,0,0};       // extra group (unused for 2-D tensor)
    __builtin_amdgcn_tensor_load_to_lds(g0, g1, g2, g3, g4, 0);
    __builtin_amdgcn_s_wait_tensorcnt(0);
  }
#else
  // Cooperative fallback: 512 threads copy 8192 x 16B chunks.
  {
    const v8bf* src = reinterpret_cast<const v8bf*>(hin);
    v8bf* dst = reinterpret_cast<v8bf*>(&hlds[0][0]);
    for (int i = threadIdx.x; i < (B_ * H_) / 8; i += 512)
      dst[i] = src[i];
  }
#endif
  __syncthreads();

  const __bf16* arow = &hlds[m * 16 + l16][0];
  const __bf16* brow = WhhT + (long long)(gate * H_ + n0 + l16) * H_;

  v8f acc = zero8();
  #pragma unroll 4
  for (int kk = 0; kk < H_; kk += 32) {
    if (kk + 256 < H_)
      __builtin_prefetch((const void*)(brow + kk + 256), 0, 1);
    v16bf a = cat8(ldv8(arow + kk + 8 * half), ldv8(arow + kk + 16 + 8 * half));
    v16bf b = cat8(ldv8(brow + kk + 8 * half), ldv8(brow + kk + 16 + 8 * half));
    acc = __builtin_amdgcn_wmma_f32_16x16x32_bf16(
        false, a, false, b, (short)0, acc, false, false);
  }

  #pragma unroll
  for (int v = 0; v < 8; ++v)
    ldsG[gate][m * 16 + v + 8 * half][l16] = acc[v];
  __syncthreads();

  // 1024 output elements (64 rows x 16 cols), 512 threads -> 2 each
  for (int e = threadIdx.x; e < B_ * 16; e += 512) {
    int row = e >> 4;
    int col = e & 15;
    int j   = n0 + col;
    const float* xp = Xpre_t + (long long)row * G4H + j;
    float gi = ldsG[0][row][col] + xp[0 * H_];
    float gf = ldsG[1][row][col] + xp[1 * H_];
    float gg = ldsG[2][row][col] + xp[2 * H_];
    float go = ldsG[3][row][col] + xp[3 * H_];
    float iv = sigmoidf_(gi);
    float fv = sigmoidf_(gf);
    float gv = tanhf(gg);
    float ov = sigmoidf_(go);
    long long sidx = (long long)row * H_ + j;
    float cnew = fv * cbuf[sidx] + iv * gv;
    float hnew = ov * tanhf(cnew);
    cbuf[sidx] = cnew;
    hout[sidx] = (__bf16)hnew;
    if (out_bf)  out_bf[((long long)t * B_ + row) * H_ + j] = (__bf16)hnew;
    if (out_f32) out_f32[((long long)row * T_ + t) * H_ + j] = hnew;
    if (is_last) { hT[sidx] = hnew; cT[sidx] = cnew; }
  }
}

extern "C" void kernel_launch(void* const* d_in, const int* in_sizes, int n_in,
                              void* d_out, int out_size, void* d_ws, size_t ws_size,
                              hipStream_t stream) {
  (void)in_sizes; (void)n_in; (void)out_size; (void)ws_size;
  const float* x    = (const float*)d_in[0];
  const float* h0   = (const float*)d_in[1];
  const float* c0   = (const float*)d_in[2];
  const float* Wih0 = (const float*)d_in[3];
  const float* Whh0 = (const float*)d_in[4];
  const float* b0   = (const float*)d_in[5];
  const float* Wih1 = (const float*)d_in[6];
  const float* Whh1 = (const float*)d_in[7];
  const float* b1   = (const float*)d_in[8];

  float* out   = (float*)d_out;
  float* out_y = out;                                   // [B,T,H]
  float* out_h = out + (size_t)B_ * T_ * H_;            // [2,B,H]
  float* out_c = out_h + (size_t)2 * B_ * H_;           // [2,B,H]

  char* wsp = (char*)d_ws;
  size_t off = 0;
  auto alloc = [&](size_t bytes) -> char* {
    char* p = wsp + off;
    off += (bytes + 255) & ~(size_t)255;
    return p;
  };
  __bf16* xbf   = (__bf16*)alloc((size_t)T_ * B_ * I_ * 2);
  __bf16* WihT0 = (__bf16*)alloc((size_t)G4H * I_ * 2);
  __bf16* WhhT0 = (__bf16*)alloc((size_t)G4H * H_ * 2);
  __bf16* WihT1 = (__bf16*)alloc((size_t)G4H * H_ * 2);
  __bf16* WhhT1 = (__bf16*)alloc((size_t)G4H * H_ * 2);
  __bf16* out0  = (__bf16*)alloc((size_t)T_ * B_ * H_ * 2);
  float*  Xpre  = (float*) alloc((size_t)T_ * B_ * G4H * 4);
  __bf16* hping = (__bf16*)alloc((size_t)B_ * H_ * 2);
  __bf16* hpong = (__bf16*)alloc((size_t)B_ * H_ * 2);
  float*  cbuf  = (float*) alloc((size_t)B_ * H_ * 4);

  // Precision prep: bf16 copies / transposed weights.
  k_convert_x <<<4096, 256, 0, stream>>>(x, xbf);
  k_transposeW<<<4096, 256, 0, stream>>>(Wih0, WihT0, I_);
  k_transposeW<<<4096, 256, 0, stream>>>(Whh0, WhhT0, H_);
  k_transposeW<<<4096, 256, 0, stream>>>(Wih1, WihT1, H_);
  k_transposeW<<<4096, 256, 0, stream>>>(Whh1, WhhT1, H_);

  dim3 gpre(G4H / 64, (T_ * B_) / 128);

  // ---- layer 0 ----
  k_gemm_pre<<<gpre, 256, 0, stream>>>(xbf, WihT0, b0, Xpre, T_ * B_, I_);
  k_init_state<<<(B_ * H_ + 255) / 256, 256, 0, stream>>>(h0, c0, hping, cbuf);
  for (int t = 0; t < T_; ++t) {
    __bf16* hin  = (t & 1) ? hpong : hping;
    __bf16* hout = (t & 1) ? hping : hpong;
    k_rec_step<<<H_ / 16, 512, 0, stream>>>(hin, hout, WhhT0,
        Xpre + (size_t)t * B_ * G4H, cbuf, out0, nullptr,
        out_h, out_c, t, (t == T_ - 1) ? 1 : 0);
  }

  // ---- layer 1 ----
  k_gemm_pre<<<gpre, 256, 0, stream>>>(out0, WihT1, b1, Xpre, T_ * B_, H_);
  k_init_state<<<(B_ * H_ + 255) / 256, 256, 0, stream>>>(
      h0 + (size_t)B_ * H_, c0 + (size_t)B_ * H_, hping, cbuf);
  for (int t = 0; t < T_; ++t) {
    __bf16* hin  = (t & 1) ? hpong : hping;
    __bf16* hout = (t & 1) ? hping : hpong;
    k_rec_step<<<H_ / 16, 512, 0, stream>>>(hin, hout, WhhT1,
        Xpre + (size_t)t * B_ * G4H, cbuf, nullptr, out_y,
        out_h + (size_t)B_ * H_, out_c + (size_t)B_ * H_, t, (t == T_ - 1) ? 1 : 0);
  }
}
